// TriangleMultiplicativeUpdate_62311385530946
// MI455X (gfx1250) — matrix operand order repaired and checked by reading, not compiled, not run on hardware
//
#include <hip/hip_runtime.h>
#include <hip/hip_bf16.h>

#define NDIM 512
#define NN   (512 * 512)
#define CZ   128
#define CH   128

typedef __attribute__((ext_vector_type(8)))  float  v8f;
typedef __attribute__((ext_vector_type(16))) __bf16 v16bf;
typedef __attribute__((ext_vector_type(8)))  __bf16 bf8v;

static __device__ __forceinline__ __bf16 to_bf16(float f) {
    unsigned u = __builtin_bit_cast(unsigned, f);
    u += 0x7fffu + ((u >> 16) & 1u);          // round to nearest even
    unsigned short h = (unsigned short)(u >> 16);
    return __builtin_bit_cast(__bf16, h);
}

// fast sigmoid: v_exp_f32 + v_rcp_f32 (avoid the IEEE-div sequence)
static __device__ __forceinline__ float sigmoidf(float x) {
    return __builtin_amdgcn_rcpf(1.0f + __expf(-x));
}

// Concatenate two 8-wide bf16 loads into one 16-wide WMMA fragment.
static __device__ __forceinline__ v16bf load_frag(const __bf16* p0, const __bf16* p1) {
    bf8v lo = *(const bf8v*)p0;
    bf8v hi = *(const bf8v*)p1;
    return __builtin_shufflevector(lo, hi, 0,1,2,3,4,5,6,7,8,9,10,11,12,13,14,15);
}

// D = A(16x128 rows from `arow`) x B(128x16 cols from `bcol`, stride 128) + acc
static __device__ __forceinline__ v8f wmma_chain128(const __bf16* arow,
                                                    const __bf16* bcol, v8f acc,
                                                    int lane) {
    const int half = lane >> 4;
#pragma unroll
    for (int k0 = 0; k0 < 128; k0 += 32) {
        // A: lane = row (m), elems 0-7: K = k0+8*half.., elems 8-15: K = k0+16+8*half..
        v16bf af = load_frag(arow + k0 + 8 * half, arow + k0 + 16 + 8 * half);
        // B: lane = col (n), elems 0-7: K = k0+8*half.., elems 8-15: K = k0+16+8*half..
        v16bf bf = load_frag(bcol + k0 + 8 * half, bcol + k0 + 16 + 8 * half);
        acc = __builtin_amdgcn_wmma_f32_16x16x32_bf16(false, af, false, bf,
                                                      (short)0, acc, false, false);
    }
    return acc;
}

// ---------------------------------------------------------------------------
// Kernel 1: convert + transpose the six 128x128 weight matrices to bf16.
// wt[mat][n*128 + k] = w[mat][k*128 + n]
// ---------------------------------------------------------------------------
__global__ void prep_w(const float* __restrict__ w_ap, const float* __restrict__ w_ag,
                       const float* __restrict__ w_bp, const float* __restrict__ w_bg,
                       const float* __restrict__ w_g,  const float* __restrict__ w_z,
                       __bf16* __restrict__ wt) {
    const float* srcs[6] = { w_ap, w_ag, w_bp, w_bg, w_g, w_z };
    const float* s = srcs[blockIdx.x];
    __bf16* dst = wt + (size_t)blockIdx.x * 16384;
    for (int idx = threadIdx.x; idx < 16384; idx += 256) {
        int n = idx >> 7, k = idx & 127;
        dst[idx] = to_bf16(s[k * 128 + n]);
    }
}

// ---------------------------------------------------------------------------
// Kernel 2: LayerNorm + 5 projections, gating, transposed a/b stores.
// One block = 128 flattened (i,j) rows. 256 threads = 8 waves.
// ---------------------------------------------------------------------------
__global__ void ln_proj(const float* __restrict__ z, const float* __restrict__ mask,
                        const float* __restrict__ ln_g, const float* __restrict__ ln_b,
                        const float* __restrict__ b_ap, const float* __restrict__ b_ag,
                        const float* __restrict__ b_bp, const float* __restrict__ b_bg,
                        const float* __restrict__ b_g,
                        const __bf16* __restrict__ wt,   // 6 transposed bf16 matrices
                        __bf16* __restrict__ a_t, __bf16* __restrict__ b_t,
                        float* __restrict__ g_buf) {
    __shared__ __bf16 zn_s[128 * 128];   // 32 KB normalized rows (row-major, K fast)
    __shared__ __bf16 stage[128 * 128];  // 32 KB transpose staging [h][row]

    const int tid  = threadIdx.x;
    const int lane = tid & 31;
    const int wave = tid >> 5;
    const int half = lane >> 4;
    const int r0   = blockIdx.x * 128;

    const __bf16* wt_ap = wt;
    const __bf16* wt_ag = wt + 16384;
    const __bf16* wt_bp = wt + 2 * 16384;
    const __bf16* wt_bg = wt + 3 * 16384;
    const __bf16* wt_g  = wt + 4 * 16384;

    // ---- LayerNorm: one thread per row (threads 0..127 active) ----
    if (tid < 128) {
        const float4* zp = (const float4*)(z + (size_t)(r0 + tid) * CZ);
        float sum = 0.f, sq = 0.f;
#pragma unroll 8
        for (int i = 0; i < 32; ++i) {
            float4 v = zp[i];
            sum += v.x + v.y + v.z + v.w;
            sq  += v.x * v.x + v.y * v.y + v.z * v.z + v.w * v.w;
        }
        float mu = sum * (1.0f / CZ);
        float var = sq * (1.0f / CZ) - mu * mu;
        float rs = rsqrtf(var + 1e-5f);
#pragma unroll 8
        for (int i = 0; i < 32; ++i) {
            float4 v = zp[i];
            int c = i * 4;
            zn_s[tid * 128 + c + 0] = to_bf16((v.x - mu) * rs * ln_g[c + 0] + ln_b[c + 0]);
            zn_s[tid * 128 + c + 1] = to_bf16((v.y - mu) * rs * ln_g[c + 1] + ln_b[c + 1]);
            zn_s[tid * 128 + c + 2] = to_bf16((v.z - mu) * rs * ln_g[c + 2] + ln_b[c + 2]);
            zn_s[tid * 128 + c + 3] = to_bf16((v.w - mu) * rs * ln_g[c + 3] + ln_b[c + 3]);
        }
    }
    __syncthreads();

    const int mcol = lane & 15;

    // ================= a = mask * sigmoid(zn@Wag+bag) * (zn@Wap+bap) ==========
    for (int t = wave; t < 64; t += 8) {
        int mt = t >> 3, nt = t & 7;
        const __bf16* arow = zn_s + (mt * 16 + mcol) * 128;
        v8f zero = {};
        v8f gacc = wmma_chain128(arow, wt_ag + (nt * 16 + mcol) * 128, zero, lane);
        v8f pacc = wmma_chain128(arow, wt_ap + (nt * 16 + mcol) * 128, zero, lane);
        int hcol = nt * 16 + mcol;
        float bg_ = b_ag[hcol], bp_ = b_ap[hcol];
#pragma unroll
        for (int e = 0; e < 8; ++e) {
            int rowl = mt * 16 + e + 8 * half;
            float mk = mask[r0 + rowl];
            float val = mk * sigmoidf(gacc[e] + bg_) * (pacc[e] + bp_);
            stage[hcol * 128 + rowl] = to_bf16(val);
        }
    }
    __syncthreads();
    for (int idx = tid; idx < 16384; idx += 256) {
        int h = idx >> 7, rr = idx & 127;
        a_t[(size_t)h * NN + r0 + rr] = stage[idx];
    }
    __syncthreads();

    // ================= b = mask * sigmoid(zn@Wbg+bbg) * (zn@Wbp+bbp) ==========
    for (int t = wave; t < 64; t += 8) {
        int mt = t >> 3, nt = t & 7;
        const __bf16* arow = zn_s + (mt * 16 + mcol) * 128;
        v8f zero = {};
        v8f gacc = wmma_chain128(arow, wt_bg + (nt * 16 + mcol) * 128, zero, lane);
        v8f pacc = wmma_chain128(arow, wt_bp + (nt * 16 + mcol) * 128, zero, lane);
        int hcol = nt * 16 + mcol;
        float bg_ = b_bg[hcol], bp_ = b_bp[hcol];
#pragma unroll
        for (int e = 0; e < 8; ++e) {
            int rowl = mt * 16 + e + 8 * half;
            float mk = mask[r0 + rowl];
            float val = mk * sigmoidf(gacc[e] + bg_) * (pacc[e] + bp_);
            stage[hcol * 128 + rowl] = to_bf16(val);
        }
    }
    __syncthreads();
    for (int idx = tid; idx < 16384; idx += 256) {
        int h = idx >> 7, rr = idx & 127;
        b_t[(size_t)h * NN + r0 + rr] = stage[idx];
    }

    // ================= g = sigmoid(zn@Wg + bg), direct fp32 store ============
    for (int t = wave; t < 64; t += 8) {
        int mt = t >> 3, nt = t & 7;
        const __bf16* arow = zn_s + (mt * 16 + mcol) * 128;
        v8f zero = {};
        v8f acc = wmma_chain128(arow, wt_g + (nt * 16 + mcol) * 128, zero, lane);
        int c = nt * 16 + mcol;
        float bb = b_g[c];
#pragma unroll
        for (int e = 0; e < 8; ++e) {
            int rowl = mt * 16 + e + 8 * half;
            g_buf[(size_t)(r0 + rowl) * CZ + c] = sigmoidf(acc[e] + bb);
        }
    }
}

// ---------------------------------------------------------------------------
// Kernel 3: triangle einsum x[i,k,h] = sum_j a[i,j,h]*b[k,j,h]
// Block = 32x32 (i,k) tile; 8 waves x 16 h-channels; 2x2 register tiling:
// 4 WMMAs per 4 fragment loads (2x data reuse, 4 independent chains).
// Output in blocked layout: x_blk[16x16 block][h][i_local*16 + k_local] (fp32).
// ---------------------------------------------------------------------------
__global__ void tri_einsum(const __bf16* __restrict__ a_t, const __bf16* __restrict__ b_t,
                           float* __restrict__ x_blk) {
    const int tid  = threadIdx.x;
    const int lane = tid & 31;
    const int wave = tid >> 5;
    const int half = lane >> 4;
    const int m    = lane & 15;
    const int i0   = blockIdx.y * 32;
    const int k0   = blockIdx.x * 32;

#pragma unroll 1
    for (int hh = 0; hh < 16; ++hh) {
        int h = wave * 16 + hh;
        const __bf16* a0 = a_t + (size_t)h * NN + (size_t)(i0 + m) * NDIM;
        const __bf16* a1 = a0 + 16 * NDIM;
        const __bf16* b0 = b_t + (size_t)h * NN + (size_t)(k0 + m) * NDIM;
        const __bf16* b1 = b0 + 16 * NDIM;
        v8f acc00 = {}, acc01 = {}, acc10 = {}, acc11 = {};
#pragma unroll 2
        for (int j0 = 0; j0 < NDIM; j0 += 32) {
            v16bf af0 = load_frag(a0 + j0 + 8 * half, a0 + j0 + 16 + 8 * half);
            v16bf af1 = load_frag(a1 + j0 + 8 * half, a1 + j0 + 16 + 8 * half);
            v16bf bf0 = load_frag(b0 + j0 + 8 * half, b0 + j0 + 16 + 8 * half);
            v16bf bf1 = load_frag(b1 + j0 + 8 * half, b1 + j0 + 16 + 8 * half);
            acc00 = __builtin_amdgcn_wmma_f32_16x16x32_bf16(false, af0, false, bf0,
                                                            (short)0, acc00, false, false);
            acc01 = __builtin_amdgcn_wmma_f32_16x16x32_bf16(false, af0, false, bf1,
                                                            (short)0, acc01, false, false);
            acc10 = __builtin_amdgcn_wmma_f32_16x16x32_bf16(false, af1, false, bf0,
                                                            (short)0, acc10, false, false);
            acc11 = __builtin_amdgcn_wmma_f32_16x16x32_bf16(false, af1, false, bf1,
                                                            (short)0, acc11, false, false);
        }
        // store each sub-tile to its 16x16 block slot
        const int bi = blockIdx.y * 2, bk = blockIdx.x * 2;
        {
            float* xp = x_blk + ((size_t)bi * 32 + bk) * (128 * 256) + (size_t)h * 256;
#pragma unroll
            for (int e = 0; e < 8; ++e) xp[(e + 8 * half) * 16 + m] = acc00[e];
        }
        {
            float* xp = x_blk + ((size_t)bi * 32 + bk + 1) * (128 * 256) + (size_t)h * 256;
#pragma unroll
            for (int e = 0; e < 8; ++e) xp[(e + 8 * half) * 16 + m] = acc01[e];
        }
        {
            float* xp = x_blk + ((size_t)(bi + 1) * 32 + bk) * (128 * 256) + (size_t)h * 256;
#pragma unroll
            for (int e = 0; e < 8; ++e) xp[(e + 8 * half) * 16 + m] = acc10[e];
        }
        {
            float* xp = x_blk + ((size_t)(bi + 1) * 32 + bk + 1) * (128 * 256) + (size_t)h * 256;
#pragma unroll
            for (int e = 0; e < 8; ++e) xp[(e + 8 * half) * 16 + m] = acc11[e];
        }
    }
}

// ---------------------------------------------------------------------------
// Kernel 4: LayerNorm over h, @ w_z + b_z, * g  ->  out (fp32 [N,N,CZ])
// Block = 16x16 (i,k) tile; thread-per-pair LN; WMMA for the projection.
// ---------------------------------------------------------------------------
__global__ void final_out(const float* __restrict__ x_blk, const float* __restrict__ g_buf,
                          const float* __restrict__ ln_og, const float* __restrict__ ln_ob,
                          const __bf16* __restrict__ wzt,  const float* __restrict__ b_z,
                          float* __restrict__ out) {
    __shared__ __bf16 xn_s[256 * 128];   // 64 KB normalized x (pair-major, h fast)

    const int tid  = threadIdx.x;
    const int lane = tid & 31;
    const int wave = tid >> 5;
    const int half = lane >> 4;
    const int i0   = blockIdx.y * 16;
    const int k0   = blockIdx.x * 16;
    const size_t blk = (size_t)blockIdx.y * 32 + blockIdx.x;

    // ---- per-(i,k)-pair LayerNorm over 128 h values ----
    const float* xp = x_blk + blk * (128 * 256) + tid;   // stride 256 over h
    float sum = 0.f, sq = 0.f;
#pragma unroll 8
    for (int h = 0; h < 128; ++h) {
        float v = xp[h * 256];
        sum += v; sq += v * v;
    }
    float mu = sum * (1.0f / 128.0f);
    float var = sq * (1.0f / 128.0f) - mu * mu;
    float rs = rsqrtf(var + 1e-5f);
#pragma unroll 8
    for (int h = 0; h < 128; ++h) {
        float v = xp[h * 256];
        xn_s[tid * 128 + h] = to_bf16((v - mu) * rs * ln_og[h] + ln_ob[h]);
    }
    __syncthreads();

    // ---- [256 x 128] @ wzt -> out, fused +b_z and *g ----
    const int mcol = lane & 15;
    for (int t = wave; t < 128; t += 8) {
        int mt = t >> 3, nt = t & 7;                 // mt: 0..15 (pairs), nt: 0..7 (cz)
        const __bf16* arow = xn_s + (mt * 16 + mcol) * 128;
        v8f zero = {};
        v8f acc = wmma_chain128(arow, wzt + (nt * 16 + mcol) * 128, zero, lane);
        int c = nt * 16 + mcol;
        float bz = b_z[c];
#pragma unroll
        for (int e = 0; e < 8; ++e) {
            int rowp = mt * 16 + e + 8 * half;       // pair index
            int gi = i0 + (rowp >> 4);
            int gk = k0 + (rowp & 15);
            size_t idx = ((size_t)gi * NDIM + gk) * CZ + c;
            out[idx] = (acc[e] + bz) * g_buf[idx];
        }
    }
}

// ---------------------------------------------------------------------------
extern "C" void kernel_launch(void* const* d_in, const int* in_sizes, int n_in,
                              void* d_out, int out_size, void* d_ws, size_t ws_size,
                              hipStream_t stream) {
    (void)in_sizes; (void)n_in; (void)out_size; (void)ws_size;
    const float* z      = (const float*)d_in[0];
    const float* mask   = (const float*)d_in[1];
    const float* ln_in_g = (const float*)d_in[2];
    const float* ln_in_b = (const float*)d_in[3];
    const float* w_ap = (const float*)d_in[4];
    const float* b_ap = (const float*)d_in[5];
    const float* w_ag = (const float*)d_in[6];
    const float* b_ag = (const float*)d_in[7];
    const float* w_bp = (const float*)d_in[8];
    const float* b_bp = (const float*)d_in[9];
    const float* w_bg = (const float*)d_in[10];
    const float* b_bg = (const float*)d_in[11];
    const float* w_g  = (const float*)d_in[12];
    const float* b_g  = (const float*)d_in[13];
    const float* ln_out_g = (const float*)d_in[14];
    const float* ln_out_b = (const float*)d_in[15];
    const float* w_z  = (const float*)d_in[16];
    const float* b_z  = (const float*)d_in[17];
    float* out = (float*)d_out;

    char* ws = (char*)d_ws;
    const size_t WT_OFF = 0;                               // 6 * 32 KB bf16 weights
    const size_t A_OFF  = 1u << 20;                        // a_t bf16 [128][512*512]
    const size_t B_OFF  = A_OFF + (size_t)128 * NN * 2;    // b_t bf16
    const size_t G_OFF  = B_OFF + (size_t)128 * NN * 2;    // g fp32 [N,N,128]
    const size_t X_OFF  = G_OFF + (size_t)NN * CZ * 4;     // x blocked fp32

    __bf16* wt    = (__bf16*)(ws + WT_OFF);
    __bf16* a_t   = (__bf16*)(ws + A_OFF);
    __bf16* b_t   = (__bf16*)(ws + B_OFF);
    float*  g_buf = (float*)(ws + G_OFF);
    float*  x_blk = (float*)(ws + X_OFF);

    prep_w<<<6, 256, 0, stream>>>(w_ap, w_ag, w_bp, w_bg, w_g, w_z, wt);

    ln_proj<<<NN / 128, 256, 0, stream>>>(z, mask, ln_in_g, ln_in_b,
                                          b_ap, b_ag, b_bp, b_bg, b_g,
                                          wt, a_t, b_t, g_buf);

    tri_einsum<<<dim3(16, 16), 256, 0, stream>>>(a_t, b_t, x_blk);

    final_out<<<dim3(32, 32), 256, 0, stream>>>(x_blk, g_buf, ln_out_g, ln_out_b,
                                                wt + 5 * 16384, b_z, out);
}
